// TinyModel_36739150250067
// MI455X (gfx1250) — matrix operand order repaired and checked by reading, not compiled, not run on hardware
//
#include <hip/hip_runtime.h>

typedef __attribute__((ext_vector_type(16))) _Float16 v16h;
typedef __attribute__((ext_vector_type(8)))  float    v8f;

#define NB    128
#define CHN   24
#define INR   64
#define HW    1024      // 32*32
#define DVEC  24576     // CHN*HW
#define NPIX  131072    // NB*HW
#define LAMR  1e-4f
#define EPSR  1e-5f

// ---------------------------------------------------------------------------
// Weight packing into B-fragment order (contiguous k, same order as A):
// Wp[(ks*NT + nt)*32 + lane][16 halves], element i of lane l at k-step ks:
//   k = ks*32 + (l>>4)*16 + i,  oc = nt*16 + (l&15)
// conv1: K padded per-tap to 32 ch -> K = 9*32 = 288 (tap = k>>5, ic = k&31,
//        ic>=24 is zero padding).  conv2: K = 9*64 = 576 (tap = k>>6, ic = k&63).
// ---------------------------------------------------------------------------
__global__ void pack_w1_k(const float* __restrict__ w1, _Float16* __restrict__ Wp) {
    int idx = blockIdx.x * 256 + threadIdx.x;           // 9*4*32*16 = 18432
    if (idx >= 18432) return;
    int i  = idx & 15;
    int l  = (idx >> 4) & 31;
    int nt = (idx >> 9) & 3;
    int ks = idx >> 11;                                 // 0..8
    int k  = ks * 32 + ((l >> 4) << 4) + i;             // < 288
    int oc = nt * 16 + (l & 15);
    int tap = k >> 5, ic = k & 31;
    float v = 0.f;
    if (ic < 24) {
        int dy = tap / 3, dx = tap - dy * 3;
        v = w1[((oc * 24 + ic) * 3 + dy) * 3 + dx];
    }
    Wp[idx] = (_Float16)v;
}

__global__ void pack_w2_k(const float* __restrict__ w2, _Float16* __restrict__ Wp) {
    int idx = blockIdx.x * 256 + threadIdx.x;           // 18*2*32*16 = 18432
    if (idx >= 18432) return;
    int i  = idx & 15;
    int l  = (idx >> 4) & 31;
    int nt = (idx >> 9) & 1;
    int ks = idx >> 10;                                 // 0..17
    int k  = ks * 32 + ((l >> 4) << 4) + i;             // < 576
    int oc = nt * 16 + (l & 15);
    float v = 0.f;
    if (oc < 24) {
        int tap = k >> 6, ic = k & 63;
        int dy = tap / 3, dx = tap - dy * 3;
        v = w2[((oc * 64 + ic) * 3 + dy) * 3 + dx];
    }
    Wp[idx] = (_Float16)v;
}

// ---------------------------------------------------------------------------
// Implicit-GEMM 3x3 SAME conv on WMMA.
// One wave -> 32 pixels (one full image row) x (NT*16) output channels.
// K in 32-steps; each step maps to exactly one (dy,dx) tap, so halo bounds
// and row addressing are computed once per step (no per-element div/mod).
// A layout: row m = lane&15; element i -> k = (lane>>4)*16 + i
// C layout: vgpr r -> m = r + (lane>>4)*8, n = lane&15
// ---------------------------------------------------------------------------
template <int CINP, int CINV, int NT, int KSTEPS, int COUTV>
__global__ void conv_wmma_k(const float* __restrict__ z, long zbstride,
                            const _Float16* __restrict__ Wp,
                            float* __restrict__ out) {
    const int COUT = NT * 16;
    int wave = (blockIdx.x * 256 + threadIdx.x) >> 5;   // 4096 waves total
    int lane = threadIdx.x & 31;
    int nlo  = lane & 15;
    int hi   = lane >> 4;
    int p0   = wave << 5;                               // 32 pixels = one row

    int bA = p0 >> 10;
    int yA = (p0 & 1023) >> 5;                          // row; x0 = 0
    const float* zrow = z + (long)bA * zbstride;

    __builtin_prefetch(Wp, 0, 1);                       // global_prefetch_b8

    v8f acc[2][NT];
#pragma unroll
    for (int mt = 0; mt < 2; ++mt)
#pragma unroll
        for (int nt = 0; nt < NT; ++nt)
#pragma unroll
            for (int e = 0; e < 8; ++e) acc[mt][nt][e] = 0.f;

    for (int ks = 0; ks < KSTEPS; ++ks) {
        int tap, icbase;
        if (CINP == 32) { tap = ks;      icbase = 0; }
        else            { tap = ks >> 1; icbase = (ks & 1) * 32; }
        int dy = tap / 3, dx = tap - dy * 3;
        int yy  = yA + dy - 1;
        int xx0 = nlo + dx - 1;                         // tile 0 column
        int xx1 = xx0 + 16;                             // tile 1 column
        bool rowok = (unsigned)yy < 32u;
        bool c0 = rowok && ((unsigned)xx0 < 32u);
        bool c1 = rowok && ((unsigned)xx1 < 32u);
        long rbase = (long)yy * 32;

        union { v16h v; _Float16 h[16]; } A0, A1;
#pragma unroll
        for (int i = 0; i < 16; ++i) {
            int ic = icbase + (hi << 4) + i;
            bool icok = (CINV == CINP) || (ic < CINV);
            long o = (long)ic * 1024 + rbase;
            float v0 = (c0 && icok) ? zrow[o + xx0] : 0.f;
            float v1 = (c1 && icok) ? zrow[o + xx1] : 0.f;
            A0.h[i] = (_Float16)v0;
            A1.h[i] = (_Float16)v1;
        }
#pragma unroll
        for (int nt = 0; nt < NT; ++nt) {
            v16h Bv = *(const v16h*)(Wp + (((ks * NT + nt) * 32 + lane) << 4));
            acc[0][nt] = __builtin_amdgcn_wmma_f32_16x16x32_f16(
                false, A0.v, false, Bv, (short)0, acc[0][nt], false, false);
            acc[1][nt] = __builtin_amdgcn_wmma_f32_16x16x32_f16(
                false, A1.v, false, Bv, (short)0, acc[1][nt], false, false);
        }
    }

#pragma unroll
    for (int mt = 0; mt < 2; ++mt)
#pragma unroll
        for (int nt = 0; nt < NT; ++nt) {
            int oc = nt * 16 + nlo;
            if (oc < COUTV) {
#pragma unroll
                for (int r = 0; r < 8; ++r) {
                    int p = p0 + mt * 16 + (hi << 3) + r;
                    int b = p >> 10, t = p & 1023;
                    out[((long)b * COUT + oc) * 1024 + t] = acc[mt][nt][r];
                }
            }
        }
}

// ---------------------------------------------------------------------------
// GroupNorm over 8 groups; optional pre-op:
//   mode bit1: v = a + b2 ; mode bit0: v = relu(v)
// One block per (batch, group). cpg = channels per group.
// ---------------------------------------------------------------------------
__global__ void gn_k(const float* __restrict__ a, long astr,
                     const float* __restrict__ b2, long b2str,
                     float* __restrict__ out, long ostr,
                     const float* __restrict__ g, const float* __restrict__ bet,
                     int cpg, int mode) {
    int blk = blockIdx.x;
    int b   = blk >> 3, grp = blk & 7;
    int tid = threadIdx.x;
    int n   = cpg * 1024;
    const float* ab = a + (long)b * astr;
    const float* bb = (mode & 2) ? (b2 + (long)b * b2str) : ab;

    float s = 0.f, s2 = 0.f;
    for (int e = tid; e < n; e += 256) {
        int c  = grp * cpg + (e >> 10);
        long o = (long)c * 1024 + (e & 1023);
        float v = ab[o];
        if (mode & 2) v += bb[o];
        if (mode & 1) v = fmaxf(v, 0.f);
        s += v; s2 += v * v;
    }
    __shared__ float r1[256], r2[256];
    r1[tid] = s; r2[tid] = s2;
    __syncthreads();
    for (int st = 128; st > 0; st >>= 1) {
        if (tid < st) { r1[tid] += r1[tid + st]; r2[tid] += r2[tid + st]; }
        __syncthreads();
    }
    __shared__ float mu_s, rstd_s;
    if (tid == 0) {
        float mu  = r1[0] / (float)n;
        float var = r2[0] / (float)n - mu * mu;
        mu_s = mu; rstd_s = rsqrtf(var + EPSR);
    }
    __syncthreads();
    float mu = mu_s, rstd = rstd_s;
    for (int e = tid; e < n; e += 256) {
        int c  = grp * cpg + (e >> 10);
        long o = (long)c * 1024 + (e & 1023);
        float v = ab[o];
        if (mode & 2) v += bb[o];
        if (mode & 1) v = fmaxf(v, 0.f);
        out[(long)b * ostr + o] = (v - mu) * rstd * g[c] + bet[c];
    }
}

// ---------------------------------------------------------------------------
// BatchNorm (training stats over N,H,W) for C=24 contiguous [b][24][1024].
// ---------------------------------------------------------------------------
__global__ void bn_stats_k(const float* __restrict__ in, float* __restrict__ stats,
                           int relu) {
    int c = blockIdx.x, tid = threadIdx.x;
    float s = 0.f, s2 = 0.f;
    for (int i = tid; i < NPIX; i += 256) {
        int b = i >> 10, t = i & 1023;
        float v = in[((long)b * 24 + c) * 1024 + t];
        if (relu) v = fmaxf(v, 0.f);
        s += v; s2 += v * v;
    }
    __shared__ float r1[256], r2[256];
    r1[tid] = s; r2[tid] = s2;
    __syncthreads();
    for (int st = 128; st > 0; st >>= 1) {
        if (tid < st) { r1[tid] += r1[tid + st]; r2[tid] += r2[tid + st]; }
        __syncthreads();
    }
    if (tid == 0) {
        float mu  = r1[0] / (float)NPIX;
        float var = r2[0] / (float)NPIX - mu * mu;
        stats[c]      = mu;
        stats[24 + c] = rsqrtf(var + EPSR);
    }
}

__global__ void bn_apply_k(const float* __restrict__ in, float* __restrict__ out,
                           const float* __restrict__ stats,
                           const float* __restrict__ g, const float* __restrict__ bet,
                           int relu) {
    long i = (long)blockIdx.x * 256 + threadIdx.x;      // NB*24*1024
    int c = (int)((i >> 10) % 24);
    float v = in[i];
    if (relu) v = fmaxf(v, 0.f);
    out[i] = (v - stats[c]) * stats[24 + c] * g[c] + bet[c];
}

// ---------------------------------------------------------------------------
// pre conv: 3->24, 3x3 SAME, with bias.  One thread per output element.
// ---------------------------------------------------------------------------
__global__ void preconv_k(const float* __restrict__ x, const float* __restrict__ w,
                          const float* __restrict__ bias, float* __restrict__ out) {
    long i = (long)blockIdx.x * 256 + threadIdx.x;      // NB*24*1024
    int t  = (int)(i & 1023);
    int oc = (int)((i >> 10) % 24);
    int b  = (int)((i >> 10) / 24);
    int y = t >> 5, xx = t & 31;
    float s = bias[oc];
#pragma unroll
    for (int ic = 0; ic < 3; ++ic)
#pragma unroll
        for (int dy = 0; dy < 3; ++dy)
#pragma unroll
            for (int dx = 0; dx < 3; ++dx) {
                int yy = y + dy - 1, xc = xx + dx - 1;
                if ((unsigned)yy < 32u && (unsigned)xc < 32u)
                    s += x[((long)b * 3 + ic) * 1024 + yy * 32 + xc] *
                         w[((oc * 3 + ic) * 3 + dy) * 3 + dx];
            }
    out[i] = s;
}

// ---------------------------------------------------------------------------
// Anderson pieces
// ---------------------------------------------------------------------------
__global__ void gram_k(const float* __restrict__ X, const float* __restrict__ F,
                       float* __restrict__ Hbuf, int n) {
    int b = blockIdx.x, tid = threadIdx.x;
    const float* Xb = X + (long)b * 5 * DVEC;
    const float* Fb = F + (long)b * 5 * DVEC;
    float acc[25];
#pragma unroll
    for (int e = 0; e < 25; ++e) acc[e] = 0.f;
    for (int j = tid; j < DVEC; j += 256) {
        float g[5];
#pragma unroll
        for (int i = 0; i < 5; ++i) {
            float gv = 0.f;
            if (i < n) gv = Fb[i * DVEC + j] - Xb[i * DVEC + j];
            g[i] = gv;
        }
#pragma unroll
        for (int i = 0; i < 5; ++i)
#pragma unroll
            for (int m = 0; m < 5; ++m) acc[i * 5 + m] += g[i] * g[m];
    }
    __shared__ float red[256];
    float* Hb = Hbuf + b * 36;
#pragma unroll
    for (int e = 0; e < 25; ++e) {
        red[tid] = acc[e];
        __syncthreads();
        for (int st = 128; st > 0; st >>= 1) {
            if (tid < st) red[tid] += red[tid + st];
            __syncthreads();
        }
        if (tid == 0) {
            int i = e / 5, m = e % 5;
            if (i < n && m < n)
                Hb[(i + 1) * 6 + (m + 1)] = red[0] + ((i == m) ? LAMR : 0.f);
        }
        __syncthreads();
    }
    if (tid == 0) {
        Hb[0] = 0.f;
        for (int i = 1; i <= n; ++i) { Hb[i] = 1.f; Hb[i * 6] = 1.f; }
    }
}

// Gauss-Jordan with partial pivoting on (n+1)x(n+1); one thread per batch.
__global__ void solve_k(const float* __restrict__ Hbuf, float* __restrict__ alpha,
                        int n) {
    int b = blockIdx.x * blockDim.x + threadIdx.x;
    if (b >= NB) return;
    int nn = n + 1;
    float A[6][7];
    const float* Hb = Hbuf + b * 36;
    for (int i = 0; i < nn; ++i) {
        for (int j = 0; j < nn; ++j) A[i][j] = Hb[i * 6 + j];
        A[i][nn] = (i == 0) ? 1.f : 0.f;
    }
    for (int col = 0; col < nn; ++col) {
        int piv = col; float best = fabsf(A[col][col]);
        for (int r = col + 1; r < nn; ++r) {
            float v = fabsf(A[r][col]);
            if (v > best) { best = v; piv = r; }
        }
        if (piv != col)
            for (int j = 0; j <= nn; ++j) {
                float tmp = A[col][j]; A[col][j] = A[piv][j]; A[piv][j] = tmp;
            }
        float d = A[col][col];
        if (fabsf(d) < 1e-20f) d = 1e-20f;
        float inv = 1.f / d;
        for (int r = 0; r < nn; ++r) {
            if (r == col) continue;
            float f = A[r][col] * inv;
            for (int j = col; j <= nn; ++j) A[r][j] -= f * A[col][j];
        }
    }
    for (int i = 0; i < n; ++i) {
        float d = A[i + 1][i + 1];
        if (fabsf(d) < 1e-20f) d = 1e-20f;
        alpha[b * 5 + i] = A[i + 1][6] / d;
    }
}

// X[:, s] = sum_i alpha_i * F[:, i]   (BETA == 1.0)
__global__ void xnew_k(float* __restrict__ X, const float* __restrict__ F,
                       const float* __restrict__ alpha, int n, int s) {
    long gid = (long)blockIdx.x * 256 + threadIdx.x;    // NB*DVEC
    int b = (int)(gid / DVEC);
    int j = (int)(gid - (long)b * DVEC);
    float sum = 0.f;
    for (int i = 0; i < n; ++i)
        sum += alpha[b * 5 + i] * F[((long)b * 5 + i) * DVEC + j];
    X[((long)b * 5 + s) * DVEC + j] = sum;
}

// X[:, 1] = F[:, 0]
__global__ void copy01_k(float* __restrict__ X, const float* __restrict__ F) {
    long gid = (long)blockIdx.x * 256 + threadIdx.x;
    int b = (int)(gid / DVEC);
    int j = (int)(gid - (long)b * DVEC);
    X[((long)b * 5 + 1) * DVEC + j] = F[((long)b * 5 + 0) * DVEC + j];
}

// ---------------------------------------------------------------------------
// avgpool(8x8) + FC.  One block per batch element.
// ---------------------------------------------------------------------------
__global__ void poolfc_k(const float* __restrict__ h, const float* __restrict__ fcw,
                         const float* __restrict__ fcb, float* __restrict__ out) {
    int b = blockIdx.x, tid = threadIdx.x;
    __shared__ float pooled[384];
    for (int idx = tid; idx < 384; idx += 256) {
        int c = idx >> 4, q = idx & 15, py = q >> 2, px = q & 3;
        float s = 0.f;
        for (int dy = 0; dy < 8; ++dy)
            for (int dx = 0; dx < 8; ++dx)
                s += h[((long)b * 24 + c) * 1024 + (py * 8 + dy) * 32 + (px * 8 + dx)];
        pooled[idx] = s * (1.f / 64.f);
    }
    __syncthreads();
    if (tid < 10) {
        float s = fcb[tid];
        for (int j = 0; j < 384; ++j) s += pooled[j] * fcw[tid * 384 + j];
        out[b * 10 + tid] = s;
    }
}

// ---------------------------------------------------------------------------
extern "C" void kernel_launch(void* const* d_in, const int* in_sizes, int n_in,
                              void* d_out, int out_size, void* d_ws, size_t ws_size,
                              hipStream_t stream) {
    const float* x_in  = (const float*)d_in[0];
    const float* pcw   = (const float*)d_in[1];
    const float* pcb   = (const float*)d_in[2];
    const float* pbg   = (const float*)d_in[3];
    const float* pbb   = (const float*)d_in[4];
    const float* w1    = (const float*)d_in[5];
    const float* g1g   = (const float*)d_in[6];
    const float* g1b   = (const float*)d_in[7];
    const float* w2    = (const float*)d_in[8];
    const float* g2g   = (const float*)d_in[9];
    const float* g2b   = (const float*)d_in[10];
    const float* g3g   = (const float*)d_in[11];
    const float* g3b   = (const float*)d_in[12];
    const float* qbg   = (const float*)d_in[13];
    const float* qbb   = (const float*)d_in[14];
    const float* fcw   = (const float*)d_in[15];
    const float* fcb   = (const float*)d_in[16];
    float* out = (float*)d_out;

    const size_t SZ24 = (size_t)NB * 24 * 1024;   // 3,145,728
    const size_t SZ64 = (size_t)NB * 64 * 1024;   // 8,388,608

    float* wsf = (float*)d_ws;
    size_t off = 0;
    auto alloc = [&](size_t n) { float* p = wsf + off; off += (n + 63) & ~(size_t)63; return p; };

    _Float16* Wp1 = (_Float16*)alloc(9216);       // 18432 halves
    _Float16* Wp2 = (_Float16*)alloc(9216);       // 18432 halves
    float* xp   = alloc(SZ24);
    float* c1   = alloc(SZ64);                    // conv1 out; also pre-conv tmp
    float* yb   = alloc(SZ64);                    // gn1 out; also post-BN out
    float* c2   = alloc(SZ24);
    float* tb   = alloc(SZ24);
    float* zb   = alloc(SZ24);
    float* X    = alloc(5 * SZ24);
    float* F    = alloc(5 * SZ24);
    float* Hbuf = alloc(NB * 36);
    float* alph = alloc(NB * 5);
    float* stat = alloc(64);

    const long SLOT = 5 * (long)DVEC;             // batch stride inside X/F
    const int  GELT = NB * DVEC / 256;            // 12288 blocks for [B,24,32,32]
    const int  CGRD = 512;                        // 4096 waves, 32 pixels each

    // One f evaluation: z(strided) -> out(strided)
    auto eval_f = [&](const float* z, long zstr, float* o, long ostr) {
        conv_wmma_k<32, 24, 4, 9, 64><<<CGRD, 256, 0, stream>>>(z, zstr, Wp1, c1);
        gn_k<<<1024, 256, 0, stream>>>(c1, 65536, nullptr, 0, yb, 65536, g1g, g1b, 8, 1);
        conv_wmma_k<64, 64, 2, 18, 24><<<CGRD, 256, 0, stream>>>(yb, 65536, Wp2, c2);
        gn_k<<<1024, 256, 0, stream>>>(c2, 24576, xp, 24576, tb, 24576, g2g, g2b, 3, 2);
        gn_k<<<1024, 256, 0, stream>>>(tb, 24576, z, zstr, o, ostr, g3g, g3b, 3, 3);
    };

    // pack weights (deterministic, cheap)
    pack_w1_k<<<72, 256, 0, stream>>>(w1, Wp1);
    pack_w2_k<<<72, 256, 0, stream>>>(w2, Wp2);

    // x0 = 0 in X slot 0 (zero the whole history region)
    hipMemsetAsync(X, 0, 10 * SZ24 * sizeof(float), stream);

    // pre: conv + batch norm -> xp
    preconv_k<<<GELT, 256, 0, stream>>>(x_in, pcw, pcb, c1);
    bn_stats_k<<<24, 256, 0, stream>>>(c1, stat, 0);
    bn_apply_k<<<GELT, 256, 0, stream>>>(c1, xp, stat, pbg, pbb, 0);

    // Anderson warmup: F0 = f(x0); X1 = F0; F1 = f(X1)
    eval_f(X + 0 * DVEC, SLOT, F + 0 * DVEC, SLOT);
    copy01_k<<<GELT, 256, 0, stream>>>(X, F);
    eval_f(X + 1 * DVEC, SLOT, F + 1 * DVEC, SLOT);

    // main loop k = 2 .. 24
    for (int k = 2; k < 25; ++k) {
        int n = (k < 5) ? k : 5;
        int s = k % 5;
        gram_k<<<NB, 256, 0, stream>>>(X, F, Hbuf, n);
        solve_k<<<1, 128, 0, stream>>>(Hbuf, alph, n);
        xnew_k<<<GELT, 256, 0, stream>>>(X, F, alph, n, s);
        eval_f(X + s * DVEC, SLOT, F + s * DVEC, SLOT);
    }

    // z = f(z_star), z_star = X[:, (25-1)%5 = 4]
    eval_f(X + 4 * DVEC, SLOT, zb, DVEC);

    // post: BN(relu(z)) -> avgpool(8x8) -> FC
    bn_stats_k<<<24, 256, 0, stream>>>(zb, stat, 1);
    bn_apply_k<<<GELT, 256, 0, stream>>>(zb, yb, stat, qbg, qbb, 1);
    poolfc_k<<<NB, 256, 0, stream>>>(yb, fcw, fcb, out);
}
